// RecurrentDiscriminator_4629974745654
// MI455X (gfx1250) — matrix operand order repaired and checked by reading, not compiled, not run on hardware
//
#include <hip/hip_runtime.h>
#include <hip/hip_bf16.h>

typedef __attribute__((ext_vector_type(16))) _Float16 v16h;
typedef __attribute__((ext_vector_type(2)))  _Float16 h2v;
typedef __attribute__((ext_vector_type(8)))  float    v8f;
typedef __attribute__((ext_vector_type(4)))  float    v4f;
typedef __attribute__((ext_vector_type(8)))  int      v8i;

#define SEQ_LEN 64
#define OBS_DIM 20
#define HONE 0x00003C00   // packed {half(0.0), half(1.0)} -> 1.0 in low half

// Pack two f32 into packed-half dword (v_cvt_pk_f16_f32).
__device__ __forceinline__ int pk2(float a, float b) {
  h2v v; v[0] = (_Float16)a; v[1] = (_Float16)b;
  return __builtin_bit_cast(int, v);
}
// Exchange dword between lane l and lane l^16 (half-wave swap), no LDS memory:
// ds_swizzle_b32 group-of-32 mode, xor=0x10, and=0x1f.
__device__ __forceinline__ int swap16_i(int x) {
  return __builtin_amdgcn_ds_swizzle(x, 0x401f);
}
__device__ __forceinline__ float swap16_f(float x) {
  return __int_as_float(swap16_i(__float_as_int(x)));
}

// Branchless tanh: CDNA5 hardware V_TANH_F32 when available.
__device__ __forceinline__ float fast_tanh(float x) {
#if __has_builtin(__builtin_amdgcn_tanhf)
  return __builtin_amdgcn_tanhf(x);
#else
  float e = __builtin_amdgcn_exp2f(x * 2.8853900817779268f);
  float r = __builtin_amdgcn_rcpf(e + 1.0f);
  return __builtin_fmaf(-2.0f, r, 1.0f);
#endif
}

__global__ __launch_bounds__(256) void recdisc_kernel(
    const float* __restrict__ states,   // [B, 64, 20]
    const float* __restrict__ actions,  // [B, 2]
    const float* __restrict__ W1, const float* __restrict__ b1,   // [32,10],[10]
    const float* __restrict__ W2, const float* __restrict__ b2,   // [10,10],[10]
    const float* __restrict__ W3, const float* __restrict__ b3,   // [10,10],[10]
    const float* __restrict__ Wg1, const float* __restrict__ bg1, // [10,10],[10]
    const float* __restrict__ Wg2, const float* __restrict__ bg2, // [10,1],[1]
    float* __restrict__ out, int B)
{
  const int lane = threadIdx.x & 31;
  const int wv   = threadIdx.x >> 5;
  const int g    = lane >> 4;   // half-wave group
  const int p    = lane & 15;   // A-row (feature) / B,D-column (batch-in-tile)
  const int tile = blockIdx.x * 8 + wv;        // 16 batch elements per wave
  if (tile * 16 >= B) return;                  // wave-uniform guard
  const int n = tile * 16 + p;                 // this lane's batch element

  // ---- constant weights in WMMA A layout (A = W^T, 16x32 f16) ----
  // lane group g, half j -> K = j + 16*g ; row M = p.
  // Bias trick: K=10 column of A2/A3/Ag carries b2/b3/bg1 (B supplies 1.0 there).
  v16h A1, A2, A3, Ag;
  #pragma unroll
  for (int j = 0; j < 16; ++j) {
    const int k = j + 16 * g;
    float a1 = 0.f, a2 = 0.f, a3 = 0.f, ag = 0.f;
    if (p < 10) {
      a1 = W1[k * 10 + p];
      if (k < 10)       { a2 = W2[k*10+p]; a3 = W3[k*10+p]; ag = Wg1[k*10+p]; }
      else if (k == 10) { a2 = b2[p];      a3 = b3[p];      ag = bg1[p]; }
    }
    A1[j] = (_Float16)a1; A2[j] = (_Float16)a2;
    A3[j] = (_Float16)a3; Ag[j] = (_Float16)ag;
  }
  // b1 broadcast in the WMMA C layout: lane holds C[m=v+8g][n] = b1[m]
  v8f cb1 = {};
  float wg2l[8];
  #pragma unroll
  for (int v = 0; v < 8; ++v) {
    const int m = v + 8 * g;
    cb1[v]  = (m < 10) ? b1[m]  : 0.f;
    wg2l[v] = (m < 10) ? Wg2[m] : 0.f;
  }
  const int apk = pk2(actions[2 * n + 0], actions[2 * n + 1]); // loop-invariant

  // state^T lives in the D/C accumulator layout: lane n holds state[v+8g][n]
  v8f st = {};
  const float* __restrict__ obs = states + (size_t)n * (SEQ_LEN * OBS_DIM);

  #pragma unroll 4
  for (int s = 0; s < SEQ_LEN; ++s) {
    const float* o = obs + s * OBS_DIM;
    __builtin_prefetch(o + 4 * OBS_DIM, 0, 3);   // speculative; safe if OOB
    const v4f o0 = *(const v4f*)(o +  0);        // obs 0..3  (80B row, 16B aligned)
    const v4f o1 = *(const v4f*)(o +  4);        // obs 4..7
    const v4f o2 = *(const v4f*)(o +  8);        // obs 8..11
    const v4f o3 = *(const v4f*)(o + 12);        // obs 12..15
    const v4f o4 = *(const v4f*)(o + 16);        // obs 16..19

    // state feats 8,9 come from partner half-wave, pre-packed as half2
    const int spk = swap16_i(pk2(st[0], st[1]));

    // B = x^T (32x16): g=0 lanes carry K=0..15, g=1 lanes K=16..31.
    // Build as 8 packed-half dwords, selected per half-wave group.
    // x = [state(10) | obs(20) | act(2)]
    v8i bx;
    bx[0] = g ? pk2(o1[2], o1[3]) : pk2(st[0], st[1]);  // k16,17 | k0,1
    bx[1] = g ? pk2(o2[0], o2[1]) : pk2(st[2], st[3]);  // k18,19 | k2,3
    bx[2] = g ? pk2(o2[2], o2[3]) : pk2(st[4], st[5]);  // k20,21 | k4,5
    bx[3] = g ? pk2(o3[0], o3[1]) : pk2(st[6], st[7]);  // k22,23 | k6,7
    bx[4] = g ? pk2(o3[2], o3[3]) : spk;                // k24,25 | k8,9
    bx[5] = g ? pk2(o4[0], o4[1]) : pk2(o0[0], o0[1]);  // k26,27 | k10,11
    bx[6] = g ? pk2(o4[2], o4[3]) : pk2(o0[2], o0[3]);  // k28,29 | k12,13
    bx[7] = g ? apk               : pk2(o1[0], o1[1]);  // k30,31 | k14,15
    const v16h Bx = __builtin_bit_cast(v16h, bx);

    // layer 1: h1^T = W1^T @ x^T + b1 (bias rides the C accumulator)
    const v8f h = __builtin_amdgcn_wmma_f32_16x16x32_f16(false, A1, false, Bx,
                                                         (short)0, cb1, false, false);
    float t[8];
    #pragma unroll
    for (int v = 0; v < 8; ++v) t[v] = fast_tanh(h[v]);   // pad rows stay 0

    // layer 2: B carries h1 in K=0..9 plus the 1.0 bias row at K=10
    const int tpk = swap16_i(pk2(t[0], t[1]));
    v8i bh;
    bh[0] = g ? 0 : pk2(t[0], t[1]);
    bh[1] = g ? 0 : pk2(t[2], t[3]);
    bh[2] = g ? 0 : pk2(t[4], t[5]);
    bh[3] = g ? 0 : pk2(t[6], t[7]);
    bh[4] = g ? 0 : tpk;            // feats 8,9 from partner
    bh[5] = g ? 0 : HONE;           // bias row k=10
    bh[6] = 0; bh[7] = 0;
    const v16h Bh = __builtin_bit_cast(v16h, bh);

    v8f z = {};
    const v8f hh = __builtin_amdgcn_wmma_f32_16x16x32_f16(false, A2, false, Bh,
                                                          (short)0, z, false, false);
    float u[8];
    #pragma unroll
    for (int v = 0; v < 8; ++v) u[v] = fast_tanh(hh[v]);  // b2 already in matmul

    // layer 3 + recurrence: state = W3^T @ h2^T + b3 + state  (C accumulator)
    const int upk = swap16_i(pk2(u[0], u[1]));
    v8i bu;
    bu[0] = g ? 0 : pk2(u[0], u[1]);
    bu[1] = g ? 0 : pk2(u[2], u[3]);
    bu[2] = g ? 0 : pk2(u[4], u[5]);
    bu[3] = g ? 0 : pk2(u[6], u[7]);
    bu[4] = g ? 0 : upk;
    bu[5] = g ? 0 : HONE;           // bias row k=10
    bu[6] = 0; bu[7] = 0;
    const v16h Bu = __builtin_bit_cast(v16h, bu);

    st = __builtin_amdgcn_wmma_f32_16x16x32_f16(false, A3, false, Bu,
                                                (short)0, st, false, false);
  }

  // ---- head: g = tanh(state @ Wg1 + bg1); out = g @ Wg2 + bg2 ----
  const int spk = swap16_i(pk2(st[0], st[1]));
  v8i bs;
  bs[0] = g ? 0 : pk2(st[0], st[1]);
  bs[1] = g ? 0 : pk2(st[2], st[3]);
  bs[2] = g ? 0 : pk2(st[4], st[5]);
  bs[3] = g ? 0 : pk2(st[6], st[7]);
  bs[4] = g ? 0 : spk;
  bs[5] = g ? 0 : HONE;             // bg1 row
  bs[6] = 0; bs[7] = 0;
  const v16h Bs = __builtin_bit_cast(v16h, bs);

  v8f z = {};
  const v8f gh = __builtin_amdgcn_wmma_f32_16x16x32_f16(false, Ag, false, Bs,
                                                        (short)0, z, false, false);
  float acc = 0.f;
  #pragma unroll
  for (int v = 0; v < 8; ++v) acc = __builtin_fmaf(fast_tanh(gh[v]), wg2l[v], acc);
  acc += swap16_f(acc);             // combine both half-waves
  if (lane < 16) out[n] = acc + bg2[0];
}

extern "C" void kernel_launch(void* const* d_in, const int* in_sizes, int n_in,
                              void* d_out, int out_size, void* d_ws, size_t ws_size,
                              hipStream_t stream) {
  (void)in_sizes; (void)n_in; (void)d_ws; (void)ws_size;
  const float* states  = (const float*)d_in[0];
  const float* actions = (const float*)d_in[1];
  const float* W1  = (const float*)d_in[2];
  const float* b1  = (const float*)d_in[3];
  const float* W2  = (const float*)d_in[4];
  const float* b2  = (const float*)d_in[5];
  const float* W3  = (const float*)d_in[6];
  const float* b3  = (const float*)d_in[7];
  const float* Wg1 = (const float*)d_in[8];
  const float* bg1 = (const float*)d_in[9];
  const float* Wg2 = (const float*)d_in[10];
  const float* bg2 = (const float*)d_in[11];
  float* out = (float*)d_out;

  const int B = out_size;                      // output is [B]
  const int tiles = (B + 15) / 16;             // 16 batch elems per wave
  const int blocks = (tiles + 7) / 8;          // 8 waves (256 thr) per block
  recdisc_kernel<<<blocks, 256, 0, stream>>>(states, actions,
                                             W1, b1, W2, b2, W3, b3,
                                             Wg1, bg1, Wg2, bg2, out, B);
}
